// SnnNetwork_1932735283499
// MI455X (gfx1250) — matrix operand order repaired and checked by reading, not compiled, not run on hardware
//
#include <hip/hip_runtime.h>
#include <hip/hip_bf16.h>
#include <math.h>

// ---------------------------------------------------------------------------
// SNN (2 recurrent adaptive-LIF layers + leaky output integrator), T=50 steps.
// bf16 WMMA (v_wmma_f32_16x16x32_bf16) for the four 256x2048x2048 GEMMs per
// step; spikes are exactly {0,1} in bf16. Weights converted/transposed to
// [K][N] bf16 once per launch (32MB, resident in 192MB L2).
// Per step: kernel A (ff1/fb1 GEMMs + layer-1 LIF), kernel B (ff2/rec2 GEMMs
// + K=10 fb2 + layer-2 LIF), kernel C (256x10x2048 GEMM + log_softmax).
// k-loop is software-pipelined (double-buffered fragments) and each wave owns
// a 16x32 output strip (2 N-tiles, 4 accumulators) so each A fragment feeds
// two WMMAs.
// ---------------------------------------------------------------------------

typedef __bf16 bf16_t;
typedef float  v8f   __attribute__((ext_vector_type(8)));
typedef bf16_t v8bf  __attribute__((ext_vector_type(8)));
typedef bf16_t v16bf __attribute__((ext_vector_type(16)));

#define HDIM   2048
#define BATCH  256
#define NOUT   10
#define TSTEPS 50      // reference T_STEPS (device scalar unreadable in capture)
#define DT_C   0.5f
#define B_J0_C 0.1f
#define BETA_C 1.8f

// ---------------- WMMA helpers (CDNA5 wave32, 16x16x32 bf16) ---------------

__device__ __forceinline__ v8f wmma_bf16(v16bf a, v16bf b, v8f c) {
  // 8 args: (neg_a, A, neg_b, B, c_mod, C, reuse_a, reuse_b)
  return __builtin_amdgcn_wmma_f32_16x16x32_bf16(false, a, false, b,
                                                 (short)0, c, false, false);
}

// A-matrix 16x32 bf16 per ISA layout: lane<16 holds M=lane, K = {k0..k0+7}
// in v0..3 halves and {k0+16..k0+23} in v4..7; lane>=16 holds K groups +8.
__device__ __forceinline__ v16bf load_a_frag(const bf16_t* __restrict__ A,
                                             int row, int k0, int half) {
  const v8bf lo = *reinterpret_cast<const v8bf*>(A + (size_t)row * HDIM + k0 + 8 * half);
  const v8bf hi = *reinterpret_cast<const v8bf*>(A + (size_t)row * HDIM + k0 + 16 + 8 * half);
  v16bf r;
#pragma unroll
  for (int j = 0; j < 8; ++j) { r[j] = lo[j]; r[j + 8] = hi[j]; }
  return r;
}

// B-matrix 32x16 bf16: lane%16 = N, K = {k..k+7} in elements 0..7 and
// {k+16..k+23} in elements 8..15 (k = k0 + 8*half).  Bt is [K][N] row-major.
__device__ __forceinline__ v16bf load_b_frag(const bf16_t* __restrict__ Bt,
                                             int k, int n0) {
  const int n = n0 + (threadIdx.x & 15);
  v16bf r;
#pragma unroll
  for (int j = 0; j < 8; ++j) {
    r[j] = Bt[(size_t)(k + j) * HDIM + n];
    r[j + 8] = Bt[(size_t)(k + 16 + j) * HDIM + n];
  }
  return r;
}

// Dual-GEMM over a 16x32 output strip: acc1{x,y} += A1 @ W1, acc2{x,y} += A2 @ W2.
// Software-pipelined: next k-tile's 4 b128 A loads and 64 bf16 B loads are in flight while the four
// WMMAs consume the previous tile's registers.
__device__ __forceinline__ void gemm_dual_strip(
    const bf16_t* __restrict__ A1, const bf16_t* __restrict__ A2,
    const bf16_t* __restrict__ W1, const bf16_t* __restrict__ W2,
    int arow, int bk, int n0, int half,
    v8f& acc1x, v8f& acc1y, v8f& acc2x, v8f& acc2y) {
  v16bf a1c  = load_a_frag(A1, arow, 0, half);
  v16bf a2c  = load_a_frag(A2, arow, 0, half);
  v16bf b1xc = load_b_frag(W1, bk, n0);
  v16bf b1yc = load_b_frag(W1, bk, n0 + 16);
  v16bf b2xc = load_b_frag(W2, bk, n0);
  v16bf b2yc = load_b_frag(W2, bk, n0 + 16);
#pragma unroll 2
  for (int k0 = 0; k0 < HDIM; k0 += 32) {
    const int kn = (k0 + 32) & (HDIM - 1);   // wraps to 0 on the last iter (harmless prefetch)
    v16bf a1n  = load_a_frag(A1, arow, kn, half);
    v16bf a2n  = load_a_frag(A2, arow, kn, half);
    v16bf b1xn = load_b_frag(W1, kn + bk, n0);
    v16bf b1yn = load_b_frag(W1, kn + bk, n0 + 16);
    v16bf b2xn = load_b_frag(W2, kn + bk, n0);
    v16bf b2yn = load_b_frag(W2, kn + bk, n0 + 16);
    acc1x = wmma_bf16(a1c, b1xc, acc1x);
    acc1y = wmma_bf16(a1c, b1yc, acc1y);
    acc2x = wmma_bf16(a2c, b2xc, acc2x);
    acc2y = wmma_bf16(a2c, b2yc, acc2y);
    a1c = a1n; a2c = a2n;
    b1xc = b1xn; b1yc = b1yn; b2xc = b2xn; b2yc = b2yn;
  }
}

// ---------------- one-time prep kernels ------------------------------------

// W [N][K] f32 row-major  ->  Bt [K][N] bf16 (tiled transpose)
__global__ __launch_bounds__(256) void transpose_bf16_kernel(
    const float* __restrict__ W, bf16_t* __restrict__ Bt) {
  __shared__ float tile[32][33];
  int kb = blockIdx.x * 32;            // K base
  int nb = blockIdx.y * 32;            // N base
  int tx = threadIdx.x & 31;
  int ty = threadIdx.x >> 5;           // 0..7
#pragma unroll
  for (int i = 0; i < 32; i += 8)
    tile[ty + i][tx] = W[(size_t)(nb + ty + i) * HDIM + kb + tx];
  __syncthreads();
#pragma unroll
  for (int i = 0; i < 32; i += 8)
    Bt[(size_t)(kb + ty + i) * HDIM + nb + tx] = (bf16_t)tile[tx][ty + i];
}

__global__ __launch_bounds__(256) void init_state_kernel(
    const float* __restrict__ x_t,
    const float* __restrict__ soma1_i, const float* __restrict__ spk1_i,
    const float* __restrict__ a1_i, const float* __restrict__ b1_i,
    const float* __restrict__ soma2_i, const float* __restrict__ spk2_i,
    const float* __restrict__ a2_i, const float* __restrict__ b2_i,
    float* __restrict__ x,
    float* __restrict__ s1, float* __restrict__ k1f, float* __restrict__ a1, float* __restrict__ b1,
    float* __restrict__ s2, float* __restrict__ k2f, float* __restrict__ a2, float* __restrict__ b2,
    bf16_t* __restrict__ k1bf0, bf16_t* __restrict__ k2bf0) {
  size_t i = (size_t)blockIdx.x * blockDim.x + threadIdx.x;
  x[i]  = x_t[i] * 0.5f;               // dropout(p=0) then *0.5 in reference
  s1[i] = soma1_i[i]; k1f[i] = spk1_i[i]; a1[i] = a1_i[i]; b1[i] = b1_i[i];
  s2[i] = soma2_i[i]; k2f[i] = spk2_i[i]; a2[i] = a2_i[i]; b2[i] = b2_i[i];
  k1bf0[i] = (bf16_t)spk1_i[i];
  k2bf0[i] = (bf16_t)spk2_i[i];
}

__global__ __launch_bounds__(256) void init_mem_kernel(
    const float* __restrict__ mem_in, float* __restrict__ mem,
    float* __restrict__ nrm) {
  int m = threadIdx.x;                 // 256 rows, 1 block
  float v[NOUT], nr = 0.f;
#pragma unroll
  for (int o = 0; o < NOUT; ++o) { v[o] = mem_in[m * NOUT + o]; nr += v[o] * v[o]; mem[m * NOUT + o] = v[o]; }
  nr = fmaxf(sqrtf(nr), 1e-12f);
#pragma unroll
  for (int o = 0; o < NOUT; ++o) nrm[m * NOUT + o] = v[o] / nr;
}

// ---------------- per-step kernel A: layer 1 -------------------------------
// acc1 = k1 @ rec_w1^T ; acc2 = k2 @ w21^T ; then adaptive-LIF update.
// 128 threads = 4 waves; each wave owns a 16x32 strip; grid (16,16).
__global__ __launch_bounds__(128) void layer1_kernel(
    const bf16_t* __restrict__ k1bf, const bf16_t* __restrict__ k2bf,
    const bf16_t* __restrict__ wt_rec1, const bf16_t* __restrict__ wt_w21,
    const float* __restrict__ x,
    const float* __restrict__ rec_b1, const float* __restrict__ b21,
    const float* __restrict__ tau_m1, const float* __restrict__ tau_adp1,
    const float* __restrict__ tau_a1,
    float* __restrict__ s1, float* __restrict__ k1f,
    float* __restrict__ a1, float* __restrict__ b1,
    bf16_t* __restrict__ k1bf_out) {
  const int wave = threadIdx.x >> 5;           // 0..3
  const int lane = threadIdx.x & 31;
  const int half = lane >> 4;
  const int m0 = blockIdx.y * 16;
  const int n0 = blockIdx.x * 128 + wave * 32; // 16x32 strip per wave
  const int arow = m0 + (lane & 15);
  const int bk   = 8 * half;

  v8f accffx = {}, accffy = {}, accfbx = {}, accfby = {};
  gemm_dual_strip(k1bf, k2bf, wt_rec1, wt_w21, arow, bk, n0, half,
                  accffx, accffy, accfbx, accfby);

#pragma unroll
  for (int tile = 0; tile < 2; ++tile) {
    const int n = n0 + 16 * tile + (lane & 15);
    const v8f accff = tile ? accffy : accffx;
    const v8f accfb = tile ? accfby : accfbx;
    const float rb    = rec_b1[n];
    const float fbb   = b21[n];
    const float alpha = __expf(-DT_C / tau_m1[n]);
    const float rho   = __expf(-DT_C / tau_adp1[n]);
    const float eta   = __expf(-DT_C / tau_a1[n]);
#pragma unroll
    for (int r = 0; r < 8; ++r) {
      const int m = m0 + r + 8 * half;
      const size_t idx = (size_t)m * HDIM + n;
      float ff  = x[idx] + accff[r] + rb;
      float fb  = accfb[r] + fbb;
      float spk = k1f[idx];
      float bn  = rho * b1[idx] + (1.f - rho) * spk;
      float thre = B_J0_C + BETA_C * bn;
      float an  = eta * a1[idx] + fb;
      float sg  = (1.f / (1.f + __expf(-an)) - 0.5f) * 0.5f;   // shifted_sigmoid
      float sn  = alpha * s1[idx] + sg + ff - thre * spk;
      float spn = (sn - thre) > 0.f ? 1.f : 0.f;
      s1[idx] = sn; a1[idx] = an; b1[idx] = bn; k1f[idx] = spn;
      k1bf_out[idx] = (bf16_t)spn;
    }
  }
}

// ---------------- per-step kernel B: layer 2 -------------------------------
// acc1 = k1_new @ w12^T ; acc2 = k2 @ rec_w2^T ; fb2 = nrm @ w_o2l2^T.
__global__ __launch_bounds__(128) void layer2_kernel(
    const bf16_t* __restrict__ k1bf_new, const bf16_t* __restrict__ k2bf,
    const bf16_t* __restrict__ wt_w12, const bf16_t* __restrict__ wt_rec2,
    const float* __restrict__ b12, const float* __restrict__ rec_b2,
    const float* __restrict__ w_o2l2, const float* __restrict__ b_o2l2,
    const float* __restrict__ nrm,
    const float* __restrict__ tau_m2, const float* __restrict__ tau_adp2,
    const float* __restrict__ tau_a2,
    float* __restrict__ s2, float* __restrict__ k2f,
    float* __restrict__ a2, float* __restrict__ b2,
    bf16_t* __restrict__ k2bf_out) {
  const int wave = threadIdx.x >> 5;
  const int lane = threadIdx.x & 31;
  const int half = lane >> 4;
  const int m0 = blockIdx.y * 16;
  const int n0 = blockIdx.x * 128 + wave * 32;
  const int arow = m0 + (lane & 15);
  const int bk   = 8 * half;

  v8f accffx = {}, accffy = {}, accrcx = {}, accrcy = {};
  gemm_dual_strip(k1bf_new, k2bf, wt_w12, wt_rec2, arow, bk, n0, half,
                  accffx, accffy, accrcx, accrcy);

#pragma unroll
  for (int tile = 0; tile < 2; ++tile) {
    const int n = n0 + 16 * tile + (lane & 15);
    const v8f accff = tile ? accffy : accffx;
    const v8f accrc = tile ? accrcy : accrcx;
    const float ffb   = b12[n] + rec_b2[n];
    const float fbb   = b_o2l2[n];
    const float alpha = __expf(-DT_C / tau_m2[n]);
    const float rho   = __expf(-DT_C / tau_adp2[n]);
    const float eta   = __expf(-DT_C / tau_a2[n]);
    float wv[NOUT];
#pragma unroll
    for (int o = 0; o < NOUT; ++o) wv[o] = w_o2l2[(size_t)n * NOUT + o];
#pragma unroll
    for (int r = 0; r < 8; ++r) {
      const int m = m0 + r + 8 * half;
      const size_t idx = (size_t)m * HDIM + n;
      float fb = fbb;
#pragma unroll
      for (int o = 0; o < NOUT; ++o) fb += nrm[m * NOUT + o] * wv[o];
      float ff  = accff[r] + accrc[r] + ffb;
      float spk = k2f[idx];
      float bn  = rho * b2[idx] + (1.f - rho) * spk;
      float thre = B_J0_C + BETA_C * bn;
      float an  = eta * a2[idx] + fb;
      float sg  = (1.f / (1.f + __expf(-an)) - 0.5f) * 0.5f;
      float sn  = alpha * s2[idx] + sg + ff - thre * spk;
      float spn = (sn - thre) > 0.f ? 1.f : 0.f;
      s2[idx] = sn; a2[idx] = an; b2[idx] = bn; k2f[idx] = spn;
      k2bf_out[idx] = (bf16_t)spn;
    }
  }
}

// ---------------- per-step kernel C: output integrator ---------------------
// mem_n = (mem + k2_new @ w_out^T + b_out) * exp(-1/tau_m_out); log_softmax.
__global__ __launch_bounds__(256) void out_kernel(
    const float* __restrict__ k2f, const float* __restrict__ w_out,
    const float* __restrict__ b_out, const float* __restrict__ tau_m_out,
    float* __restrict__ mem, float* __restrict__ nrm,
    float* __restrict__ out_t) {
  __shared__ float red[NOUT * 256];
  const int m = blockIdx.x;
  const int tid = threadIdx.x;
  float part[NOUT];
#pragma unroll
  for (int o = 0; o < NOUT; ++o) part[o] = 0.f;
  for (int i = tid; i < HDIM; i += 256) {
    float v = k2f[(size_t)m * HDIM + i];
#pragma unroll
    for (int o = 0; o < NOUT; ++o) part[o] += v * w_out[(size_t)o * HDIM + i];
  }
#pragma unroll
  for (int o = 0; o < NOUT; ++o) red[o * 256 + tid] = part[o];
  __syncthreads();
  for (int s = 128; s > 0; s >>= 1) {
    if (tid < s) {
#pragma unroll
      for (int o = 0; o < NOUT; ++o) red[o * 256 + tid] += red[o * 256 + tid + s];
    }
    __syncthreads();
  }
  if (tid == 0) {
    float mn[NOUT], mx = -INFINITY, ss = 0.f, nr = 0.f;
#pragma unroll
    for (int o = 0; o < NOUT; ++o) {
      float a_out = __expf(-1.0f / tau_m_out[o]);
      mn[o] = (mem[m * NOUT + o] + red[o * 256] + b_out[o]) * a_out;
      mx = fmaxf(mx, mn[o]);
    }
#pragma unroll
    for (int o = 0; o < NOUT; ++o) ss += __expf(mn[o] - mx);
    float lse = mx + __logf(ss);
#pragma unroll
    for (int o = 0; o < NOUT; ++o) {
      out_t[m * NOUT + o] = mn[o] - lse;
      mem[m * NOUT + o] = mn[o];
      nr += mn[o] * mn[o];
    }
    nr = fmaxf(sqrtf(nr), 1e-12f);
#pragma unroll
    for (int o = 0; o < NOUT; ++o) nrm[m * NOUT + o] = mn[o] / nr;
  }
}

// ---------------- launcher --------------------------------------------------

extern "C" void kernel_launch(void* const* d_in, const int* in_sizes, int n_in,
                              void* d_out, int out_size, void* d_ws, size_t ws_size,
                              hipStream_t stream) {
  (void)in_sizes; (void)n_in; (void)out_size; (void)ws_size;
  const float* x_t     = (const float*)d_in[0];
  const float* soma1_i = (const float*)d_in[1];
  const float* spk1_i  = (const float*)d_in[2];
  const float* a1_i    = (const float*)d_in[3];
  const float* b1_i    = (const float*)d_in[4];
  const float* soma2_i = (const float*)d_in[5];
  const float* spk2_i  = (const float*)d_in[6];
  const float* a2_i    = (const float*)d_in[7];
  const float* b2_i    = (const float*)d_in[8];
  const float* mem_i   = (const float*)d_in[9];
  /* d_in[10] = T (device scalar; reference fixes T=50, compiled in) */
  const float* rec_w1  = (const float*)d_in[11];
  const float* rec_b1  = (const float*)d_in[12];
  const float* rec_w2  = (const float*)d_in[13];
  const float* rec_b2  = (const float*)d_in[14];
  const float* w12     = (const float*)d_in[15];
  const float* b12     = (const float*)d_in[16];
  const float* w21     = (const float*)d_in[17];
  const float* b21     = (const float*)d_in[18];
  const float* w_out   = (const float*)d_in[19];
  const float* b_out   = (const float*)d_in[20];
  const float* w_o2l2  = (const float*)d_in[21];
  const float* b_o2l2  = (const float*)d_in[22];
  const float* tau_adp1 = (const float*)d_in[23];
  const float* tau_m1   = (const float*)d_in[24];
  const float* tau_a1   = (const float*)d_in[25];
  const float* tau_adp2 = (const float*)d_in[26];
  const float* tau_m2   = (const float*)d_in[27];
  const float* tau_a2   = (const float*)d_in[28];
  const float* tau_m_out = (const float*)d_in[29];
  float* out = (float*)d_out;

  // workspace bump allocator
  char* ws = (char*)d_ws;
  size_t off = 0;
  auto alloc = [&](size_t bytes) -> void* {
    void* p = ws + off;
    off = (off + bytes + 255) & ~(size_t)255;
    return p;
  };
  const size_t HH = (size_t)HDIM * HDIM;
  const size_t BH = (size_t)BATCH * HDIM;
  bf16_t* wt_rec1 = (bf16_t*)alloc(HH * 2);
  bf16_t* wt_rec2 = (bf16_t*)alloc(HH * 2);
  bf16_t* wt_w12  = (bf16_t*)alloc(HH * 2);
  bf16_t* wt_w21  = (bf16_t*)alloc(HH * 2);
  float*  x   = (float*)alloc(BH * 4);
  float*  s1  = (float*)alloc(BH * 4);
  float*  k1f = (float*)alloc(BH * 4);
  float*  a1  = (float*)alloc(BH * 4);
  float*  b1  = (float*)alloc(BH * 4);
  float*  s2  = (float*)alloc(BH * 4);
  float*  k2f = (float*)alloc(BH * 4);
  float*  a2  = (float*)alloc(BH * 4);
  float*  b2  = (float*)alloc(BH * 4);
  bf16_t* k1bf[2] = {(bf16_t*)alloc(BH * 2), (bf16_t*)alloc(BH * 2)};
  bf16_t* k2bf[2] = {(bf16_t*)alloc(BH * 2), (bf16_t*)alloc(BH * 2)};
  float*  mem = (float*)alloc((size_t)BATCH * NOUT * 4);
  float*  nrm = (float*)alloc((size_t)BATCH * NOUT * 4);

  // 1) weight prep: f32 [N][K] -> bf16 [K][N]
  dim3 tg(HDIM / 32, HDIM / 32);
  transpose_bf16_kernel<<<tg, 256, 0, stream>>>(rec_w1, wt_rec1);
  transpose_bf16_kernel<<<tg, 256, 0, stream>>>(rec_w2, wt_rec2);
  transpose_bf16_kernel<<<tg, 256, 0, stream>>>(w12,    wt_w12);
  transpose_bf16_kernel<<<tg, 256, 0, stream>>>(w21,    wt_w21);

  // 2) state init
  init_state_kernel<<<BH / 256, 256, 0, stream>>>(
      x_t, soma1_i, spk1_i, a1_i, b1_i, soma2_i, spk2_i, a2_i, b2_i,
      x, s1, k1f, a1, b1, s2, k2f, a2, b2, k1bf[0], k2bf[0]);
  init_mem_kernel<<<1, 256, 0, stream>>>(mem_i, mem, nrm);

  // 3) timestep loop: 4 waves/block, each wave owns a 16x32 strip,
  //    grid = (2048/128 N-strips, 256/16 M-tiles) = (16,16)
  dim3 gg(HDIM / 128, BATCH / 16);
  for (int t = 0; t < TSTEPS; ++t) {
    int cur = t & 1, nxt = cur ^ 1;
    layer1_kernel<<<gg, 128, 0, stream>>>(
        k1bf[cur], k2bf[cur], wt_rec1, wt_w21, x,
        rec_b1, b21, tau_m1, tau_adp1, tau_a1,
        s1, k1f, a1, b1, k1bf[nxt]);
    layer2_kernel<<<gg, 128, 0, stream>>>(
        k1bf[nxt], k2bf[cur], wt_w12, wt_rec2,
        b12, rec_b2, w_o2l2, b_o2l2, nrm,
        tau_m2, tau_adp2, tau_a2,
        s2, k2f, a2, b2, k2bf[nxt]);
    out_kernel<<<BATCH, 256, 0, stream>>>(
        k2f, w_out, b_out, tau_m_out, mem, nrm,
        out + (size_t)t * BATCH * NOUT);
  }
}
